// HypAttention_80169859547180
// MI455X (gfx1250) — matrix operand order repaired and checked by reading, not compiled, not run on hardware
//
#include <hip/hip_runtime.h>

#define BB 2
#define NN 4096
#define CC 512
#define HH 8
#define DD 64

typedef __attribute__((ext_vector_type(16))) __bf16 v16bf;
typedef __attribute__((ext_vector_type(8)))  __bf16 v8bf;
typedef __attribute__((ext_vector_type(8)))  float  v8f;
typedef unsigned short ushort_t;

#if defined(__AMDGCN__) && __has_builtin(__builtin_amdgcn_global_load_async_to_lds_b128) && __has_builtin(__builtin_amdgcn_s_wait_asynccnt)
#define USE_ASYNC 1
typedef __attribute__((__vector_size__(16))) int v4i_t;               // b128 payload type
typedef __attribute__((address_space(3))) v4i_t lds_v4i;              // LDS-side pointer
#else
#define USE_ASYNC 0
#endif

__device__ __forceinline__ ushort_t f2bf(float f) {
  union { float f; unsigned u; } x; x.f = f;
  unsigned r = x.u + 0x7fffu + ((x.u >> 16) & 1u);
  return (ushort_t)(r >> 16);
}

__device__ __forceinline__ v16bf cat8(v8bf lo, v8bf hi) {
  return __builtin_shufflevector(lo, hi, 0,1,2,3,4,5,6,7,8,9,10,11,12,13,14,15);
}

__device__ __forceinline__ v8f wmma_bf(v16bf a, v16bf b, v8f c) {
  return __builtin_amdgcn_wmma_f32_16x16x32_bf16(false, a, false, b, (short)0, c, false, false);
}

// ---------------- conversion kernels ----------------
__global__ void k_cvt(const float* __restrict__ s, ushort_t* __restrict__ d, int n) {
  int i = blockIdx.x * blockDim.x + threadIdx.x;
  if (i < n) d[i] = f2bf(s[i]);
}

// Mt[e*64+d] = M[d*64+e] (bf16)
__global__ void k_cvtM(const float* __restrict__ M, ushort_t* __restrict__ Mt) {
  for (int i = threadIdx.x; i < DD * DD; i += blockDim.x) {
    int e = i >> 6, d = i & 63;
    Mt[i] = f2bf(M[d * DD + e]);
  }
}

// ---------------- linear: hp = h @ W^T + b, packed bf16 into [B,H,N,D] ----------------
__global__ void k_linear(const ushort_t* __restrict__ h_bf,
                         const ushort_t* __restrict__ W_bf,
                         const float* __restrict__ bias,
                         ushort_t* __restrict__ kv_bf) {
  const int lane = threadIdx.x & 31;
  const int wave = threadIdx.x >> 5;      // = head
  const int lane16 = lane & 15;
  const int half = lane >> 4;
  const int head = wave;
  const int row_base = blockIdx.x * 16;   // flat over B*N

  v8f acc[4];
#pragma unroll
  for (int t = 0; t < 4; ++t) acc[t] = (v8f){};

  const int arow = row_base + lane16;
#pragma unroll 4
  for (int kk = 0; kk < CC / 32; ++kk) {
    const ushort_t* ap = h_bf + (size_t)arow * CC + kk * 32;
    v16bf a = *(const v16bf*)(ap + half * 16);
#pragma unroll
    for (int t = 0; t < 4; ++t) {
      int o = head * DD + t * 16 + lane16;
      v16bf b = *(const v16bf*)(W_bf + (size_t)o * CC + kk * 32 + half * 16);
      acc[t] = wmma_bf(a, b, acc[t]);
    }
  }

#pragma unroll
  for (int t = 0; t < 4; ++t) {
    int o = head * DD + t * 16 + lane16;
    float bv = bias[o];
#pragma unroll
    for (int v = 0; v < 8; ++v) {
      int row = row_base + v + 8 * half;       // flat over B*N
      int b_i = row >> 12;                     // /4096
      int n = row & (NN - 1);
      float val = acc[t][v] + bv;
      kv_bf[(((size_t)b_i * HH + head) * NN + n) * DD + t * 16 + lane16] = f2bf(val);
    }
  }
}

// ---------------- Q = hp @ M per head ----------------
__global__ void k_qgemm(const ushort_t* __restrict__ kv_bf,
                        const ushort_t* __restrict__ Mt_bf,
                        ushort_t* __restrict__ q_bf) {
  const int lane = threadIdx.x & 31;
  const int wave = threadIdx.x >> 5;
  const int lane16 = lane & 15;
  const int half = lane >> 4;
  const int tile = blockIdx.x * 8 + wave;   // 4096 tiles
  const int bh = tile >> 8;                 // /256
  const int nt = tile & 255;

  const ushort_t* src = kv_bf + ((size_t)bh * NN + nt * 16 + lane16) * DD;
  v16bf a0 = *(const v16bf*)(src + half * 16);
  v16bf a1 = *(const v16bf*)(src + 32 + half * 16);

#pragma unroll
  for (int t = 0; t < 4; ++t) {
    v16bf b0 = *(const v16bf*)(Mt_bf + (size_t)(t * 16 + lane16) * DD + half * 16);
    v16bf b1 = *(const v16bf*)(Mt_bf + (size_t)(t * 16 + lane16) * DD + 32 + half * 16);
    v8f c = (v8f){};
    c = wmma_bf(a0, b0, c);
    c = wmma_bf(a1, b1, c);
#pragma unroll
    for (int v = 0; v < 8; ++v) {
      q_bf[((size_t)bh * NN + nt * 16 + v + 8 * half) * DD + t * 16 + lane16] = f2bf(c[v]);
    }
  }
}

// ---------------- flash attention: O = softmax(Q K^T) V ----------------
__global__ void k_attn(const ushort_t* __restrict__ q_bf,
                       const ushort_t* __restrict__ kv_bf,
                       float* __restrict__ o_ws) {
  __shared__ ushort_t kLds[2][32 * DD];     // double-buffered [key][d]
  __shared__ ushort_t vtLds[DD * 32];       // [d][key]
  __shared__ ushort_t pLds[8][16 * 32];     // per-wave P staging

  const int tid = threadIdx.x;
  const int lane = tid & 31;
  const int wave = tid >> 5;
  const int lane16 = lane & 15;
  const int half = lane >> 4;

  const int bh = blockIdx.x >> 5;           // /32 : which (b,h)
  const int rt = blockIdx.x & 31;           // row block of 128
  const int row_base = rt * 128 + wave * 16;

  const ushort_t* qp = q_bf + (size_t)bh * NN * DD;
  const ushort_t* kvp = kv_bf + (size_t)bh * NN * DD;

  // Q fragments (16 rows x 64)
  const ushort_t* qrow = qp + (size_t)(row_base + lane16) * DD;
  v16bf qa0 = *(const v16bf*)(qrow + half * 16);
  v16bf qa1 = *(const v16bf*)(qrow + 32 + half * 16);

  float mrow[8], lrow[8];
  v8f o[4];
#pragma unroll
  for (int v = 0; v < 8; ++v) { mrow[v] = -3.0e38f; lrow[v] = 0.0f; }
#pragma unroll
  for (int t = 0; t < 4; ++t) o[t] = (v8f){};

#if USE_ASYNC
  // prologue: stream tile 0 straight into LDS (16B per thread, 4KB total)
  __builtin_amdgcn_global_load_async_to_lds_b128(
      (v4i_t*)(kvp + (size_t)tid * 8),
      (lds_v4i*)&kLds[0][tid * 8], 0, 0);
#endif

  for (int jt = 0; jt < NN / 32; ++jt) {
    const int cur = jt & 1;

#if USE_ASYNC
    __builtin_amdgcn_s_wait_asynccnt(0);    // this wave's tile slice arrived
    __syncthreads();                        // whole tile visible; prev compute done
    if (jt + 1 < NN / 32) {                 // stream next tile while we compute
      __builtin_amdgcn_global_load_async_to_lds_b128(
          (v4i_t*)(kvp + (size_t)(jt + 1) * 32 * DD + tid * 8),
          (lds_v4i*)&kLds[cur ^ 1][tid * 8], 0, 0);
    }
    {// build transposed V copy from the K tile already in LDS
      int k = tid >> 3;
      int d0 = (tid & 7) * 8;
      union { uint4 q; ushort_t s[8]; } u;
      u.q = *(const uint4*)&kLds[cur][tid * 8];
#pragma unroll
      for (int j = 0; j < 8; ++j) vtLds[(d0 + j) * 32 + k] = u.s[j];
    }
    __syncthreads();
#else
    {// synchronous fallback: global -> VGPR -> LDS (row-major + transposed)
      int k = tid >> 3;
      int d0 = (tid & 7) * 8;
      union { uint4 q; ushort_t s[8]; } u;
      u.q = *(const uint4*)(kvp + (size_t)(jt * 32 + k) * DD + d0);
      *(uint4*)&kLds[cur][k * DD + d0] = u.q;
#pragma unroll
      for (int j = 0; j < 8; ++j) vtLds[(d0 + j) * 32 + k] = u.s[j];
    }
    __syncthreads();
#endif

    // S = Q K^T for two 16-key subtiles
    v8f s[2];
#pragma unroll
    for (int sb = 0; sb < 2; ++sb) {
      v16bf kb0 = *(const v16bf*)&kLds[cur][(sb * 16 + lane16) * DD + half * 16];
      v16bf kb1 = *(const v16bf*)&kLds[cur][(sb * 16 + lane16) * DD + 32 + half * 16];
      v8f c = (v8f){};
      c = wmma_bf(qa0, kb0, c);
      c = wmma_bf(qa1, kb1, c);
      s[sb] = c;
    }

    // online softmax update (stats per row v + 8*half, shared across half-wave)
#pragma unroll
    for (int v = 0; v < 8; ++v) {
      float mx = fmaxf(s[0][v], s[1][v]);
#pragma unroll
      for (int off = 8; off >= 1; off >>= 1)
        mx = fmaxf(mx, __shfl_xor(mx, off, 32));
      float mnew = fmaxf(mrow[v], mx);
      float scale = __expf(mrow[v] - mnew);
      mrow[v] = mnew;
      float p0 = __expf(s[0][v] - mnew);
      float p1 = __expf(s[1][v] - mnew);
      float rs = p0 + p1;
#pragma unroll
      for (int off = 8; off >= 1; off >>= 1)
        rs += __shfl_xor(rs, off, 32);
      lrow[v] = lrow[v] * scale + rs;
#pragma unroll
      for (int t = 0; t < 4; ++t) o[t][v] *= scale;
      int prow = v + 8 * half;
      pLds[wave][prow * 32 + lane16] = f2bf(p0);
      pLds[wave][prow * 32 + 16 + lane16] = f2bf(p1);
    }

    // P A-fragment (16 rows x 32 keys) from per-wave LDS staging
    v16bf pa = *(const v16bf*)&pLds[wave][lane16 * 32 + half * 16];

    // O += P V
#pragma unroll
    for (int t = 0; t < 4; ++t) {
      v16bf vb = *(const v16bf*)&vtLds[(t * 16 + lane16) * 32 + half * 16];
      o[t] = wmma_bf(pa, vb, o[t]);
    }
#if !USE_ASYNC
    __syncthreads();
#endif
  }

  // finalize: O /= l ; store f32 to [B,N,C] workspace
  const int b_i = bh >> 3;
  const int head = bh & 7;
#pragma unroll
  for (int v = 0; v < 8; ++v) {
    float inv_l = 1.0f / lrow[v];
    int n = row_base + v + 8 * half;
#pragma unroll
    for (int t = 0; t < 4; ++t) {
      o_ws[((size_t)b_i * NN + n) * CC + head * DD + t * 16 + lane16] = o[t][v] * inv_l;
    }
  }
}

// ---------------- LayerNorm + ReLU, one wave per row ----------------
__global__ void k_ln(const float* __restrict__ x,
                     const float* __restrict__ gamma,
                     const float* __restrict__ beta,
                     float* __restrict__ out) {
  const int lane = threadIdx.x & 31;
  const int wave = threadIdx.x >> 5;
  const int row = blockIdx.x * 8 + wave;    // 8192 rows
  const float* xr = x + (size_t)row * CC;

  float sum = 0.0f, sq = 0.0f;
#pragma unroll
  for (int i = 0; i < CC / 32; ++i) {
    float v = xr[lane + i * 32];
    sum += v; sq += v * v;
  }
#pragma unroll
  for (int off = 16; off >= 1; off >>= 1) {
    sum += __shfl_xor(sum, off, 32);
    sq += __shfl_xor(sq, off, 32);
  }
  float mu = sum * (1.0f / CC);
  float var = sq * (1.0f / CC) - mu * mu;
  float rstd = rsqrtf(var + 1e-5f);
#pragma unroll
  for (int i = 0; i < CC / 32; ++i) {
    int c = lane + i * 32;
    float v = (xr[c] - mu) * rstd * gamma[c] + beta[c];
    out[(size_t)row * CC + c] = fmaxf(v, 0.0f);
  }
}

extern "C" void kernel_launch(void* const* d_in, const int* in_sizes, int n_in,
                              void* d_out, int out_size, void* d_ws, size_t ws_size,
                              hipStream_t stream) {
  const float* h     = (const float*)d_in[0];
  const float* W     = (const float*)d_in[1];
  const float* bias  = (const float*)d_in[2];
  const float* M     = (const float*)d_in[3];
  const float* gamma = (const float*)d_in[4];
  const float* beta  = (const float*)d_in[5];
  float* out = (float*)d_out;

  char* ws = (char*)d_ws;
  size_t off = 0;
  ushort_t* h_bf  = (ushort_t*)(ws + off); off += (size_t)BB * NN * CC * 2;   // 8 MB
  ushort_t* W_bf  = (ushort_t*)(ws + off); off += (size_t)CC * CC * 2;        // 0.5 MB
  ushort_t* Mt_bf = (ushort_t*)(ws + off); off += (size_t)DD * DD * 2;        // 8 KB
  off = (off + 255) & ~(size_t)255;
  ushort_t* kv_bf = (ushort_t*)(ws + off); off += (size_t)BB * HH * NN * DD * 2;
  ushort_t* q_bf  = (ushort_t*)(ws + off); off += (size_t)BB * HH * NN * DD * 2;
  off = (off + 255) & ~(size_t)255;
  float* o_ws = (float*)(ws + off);

  k_cvt<<<(BB * NN * CC + 255) / 256, 256, 0, stream>>>(h, h_bf, BB * NN * CC);
  k_cvt<<<(CC * CC + 255) / 256, 256, 0, stream>>>(W, W_bf, CC * CC);
  k_cvtM<<<1, 256, 0, stream>>>(M, Mt_bf);
  k_linear<<<(BB * NN) / 16, 256, 0, stream>>>(h_bf, W_bf, bias, kv_bf);
  k_qgemm<<<(BB * HH * NN / 16) / 8, 256, 0, stream>>>(kv_bf, Mt_bf, q_bf);
  k_attn<<<BB * HH * (NN / 128), 256, 0, stream>>>(q_bf, kv_bf, o_ws);
  k_ln<<<(BB * NN) / 8, 256, 0, stream>>>(o_ws, gamma, beta, out);
}
